// MambaLayer_60868276518953
// MI455X (gfx1250) — matrix operand order repaired and checked by reading, not compiled, not run on hardware
//
#include <hip/hip_runtime.h>
#include <math.h>

typedef __attribute__((ext_vector_type(16))) _Float16 v16h;
typedef __attribute__((ext_vector_type(8)))  _Float16 v8h;
typedef __attribute__((ext_vector_type(8)))  float    v8f;

#define BATCH_   2
#define SEQLEN_  1024
#define DMODEL_  1024
#define DINNER_  2048
#define NHEADS_  32
#define HEADDIM_ 64
#define DSTATE_  128
#define CONVDIM_ 2304      // DINNER + 2*DSTATE
#define DINPROJ_ 4384      // 2*DINNER + 2*DSTATE + NHEADS
#define NPAD_    4416      // pad DINPROJ to multiple of 64 for 32x64 GEMM tiles
#define TOK_     (BATCH_*SEQLEN_)

#if defined(__HIP_DEVICE_COMPILE__) && \
    __has_builtin(__builtin_amdgcn_global_load_async_to_lds_b32) && \
    __has_builtin(__builtin_amdgcn_s_wait_asynccnt)
#define ASYNC_LDS_ 1
#endif

__device__ __forceinline__ float silu_f(float v) { return v / (1.0f + expf(-v)); }

// ---------------- casts ----------------
__global__ void cast_f32_to_f16(const float* __restrict__ in, _Float16* __restrict__ o, int n) {
  int i = blockIdx.x * blockDim.x + threadIdx.x;
  if (i < n) o[i] = (_Float16)in[i];
}

__global__ void cast_w_pad(const float* __restrict__ in, _Float16* __restrict__ o,
                           int rows_src, int rows_dst, int cols) {
  int i = blockIdx.x * blockDim.x + threadIdx.x;
  if (i >= rows_dst * cols) return;
  int r = i / cols;
  int c = i - r * cols;
  o[i] = (r < rows_src) ? (_Float16)in[(size_t)r * cols + c] : (_Float16)0.0f;
}

// ---------------- f16 WMMA GEMM: C[M,N] = A[M,K] * W[N,K]^T (+ residual) ----------------
// One wave per block computes a 32(M) x 64(N) tile: 2 A-frags x 4 B-frags -> 8 WMMAs
// per k-step of 32, with one-step register prefetch so fragment loads overlap WMMA.
// A-frag layout (16-bit A 16x32): lane = m + 16*khalf,
//   elems 0..15 = K[khalf*16 .. +15] (same k order as B)
// B-frag layout (16-bit B 32x16): lane = n + 16*khalf, elems 0..15 = K[khalf*16 .. +15]
// C/D f32 16x16: VGPR r -> row r + 8*khalf, col = lane & 15
struct Frags {
  union { v16h v; v8h h[2]; } a[2];
  v16h b[4];
};

__global__ void gemm_f16_wmma(const _Float16* __restrict__ A,
                              const _Float16* __restrict__ W,
                              float* __restrict__ C,
                              const float* __restrict__ residual,
                              int K, int ldc) {
  const int lane  = threadIdx.x & 31;
  const int n0    = blockIdx.x * 64;
  const int m0    = blockIdx.y * 32;
  const int r16   = lane & 15;
  const int khalf = lane >> 4;

  const _Float16* pa0 = A + (size_t)(m0 + r16) * K + khalf * 16;
  const _Float16* pa1 = pa0 + (size_t)16 * K;
  const _Float16* pb  = W + (size_t)(n0 + r16) * K + khalf * 16;
  const size_t bstep  = (size_t)16 * K;

  v8f acc[2][4] = {};
  Frags cur, nxt;

#define LOADF_(F, kk)                                                        \
  do {                                                                       \
    F.a[0].h[0] = *(const v8h*)(pa0 + (kk));                                 \
    F.a[0].h[1] = *(const v8h*)(pa0 + (kk) + 8);                             \
    F.a[1].h[0] = *(const v8h*)(pa1 + (kk));                                 \
    F.a[1].h[1] = *(const v8h*)(pa1 + (kk) + 8);                             \
    F.b[0] = *(const v16h*)(pb + (kk));                                      \
    F.b[1] = *(const v16h*)(pb + bstep + (kk));                              \
    F.b[2] = *(const v16h*)(pb + 2 * bstep + (kk));                          \
    F.b[3] = *(const v16h*)(pb + 3 * bstep + (kk));                          \
  } while (0)

  LOADF_(cur, 0);
#pragma unroll 2
  for (int k0 = 0; k0 < K; k0 += 32) {
    const int kn = (k0 + 32 < K) ? (k0 + 32) : 0;   // last prefetch is a harmless dummy
    LOADF_(nxt, kn);
#pragma unroll
    for (int mi = 0; mi < 2; ++mi)
#pragma unroll
      for (int ni = 0; ni < 4; ++ni)
        acc[mi][ni] = __builtin_amdgcn_wmma_f32_16x16x32_f16(
            false, cur.a[mi].v, false, cur.b[ni], 0, acc[mi][ni], false, false);
    cur = nxt;
  }
#undef LOADF_

#pragma unroll
  for (int mi = 0; mi < 2; ++mi) {
    const int mb = m0 + mi * 16 + khalf * 8;
#pragma unroll
    for (int r = 0; r < 8; ++r) {
      const size_t base = (size_t)(mb + r) * ldc + n0 + r16;
#pragma unroll
      for (int ni = 0; ni < 4; ++ni) {
        float v = acc[mi][ni][r];
        if (residual) v += residual[base + ni * 16];
        C[base + ni * 16] = v;
      }
    }
  }
}

// ---------------- dt = softplus(raw + bias); dA = exp(dt * -exp(A_log)) ----------------
__global__ void dt_kernel(const float* __restrict__ zx, const float* __restrict__ dt_bias,
                          const float* __restrict__ A_log,
                          float* __restrict__ dts, float* __restrict__ dAb) {
  int i = blockIdx.x * blockDim.x + threadIdx.x;  // token*NHEADS + h
  if (i >= TOK_ * NHEADS_) return;
  int t = i >> 5, h = i & 31;
  float v = zx[(size_t)t * NPAD_ + DINNER_ + CONVDIM_ + h] + dt_bias[h];
  float sp = (v > 20.f) ? v : log1pf(expf(v));
  dts[i] = sp;
  dAb[i] = expf(sp * (-expf(A_log[h])));
}

// ---------------- causal depthwise conv (width 4) + bias + SiLU ----------------
__global__ void conv_kernel(const float* __restrict__ zx, const float* __restrict__ cw,
                            const float* __restrict__ cb, float* __restrict__ co) {
  int i = blockIdx.x * blockDim.x + threadIdx.x;  // (b*L+l)*CONVDIM + c
  if (i >= TOK_ * CONVDIM_) return;
  int c = i % CONVDIM_;
  int t = i / CONVDIM_;
  int l = t & (SEQLEN_ - 1);
  float acc = cb[c];
#pragma unroll
  for (int j = 0; j < 4; ++j) {
    int lj = l - 3 + j;
    if (lj >= 0) acc += zx[(size_t)(t - 3 + j) * NPAD_ + DINNER_ + c] * cw[c * 4 + j];
  }
  co[i] = silu_f(acc);
}

// ---------------- selective scan: one block per (batch, head) ----------------
// 256 threads: p = tid>>2 (0..63), ng = tid&3; each thread holds h[p][ng*32..+31] in regs.
// Per-step B/C/x rows staged global->LDS via CDNA5 async loads (ASYNCcnt) when available.
#ifdef ASYNC_LDS_
__device__ __forceinline__ void async_g2l_b32(const float* g, float* l) {
  // flat->as1 keeps the address; flat LDS address truncated to 32 bits is the LDS offset
  __builtin_amdgcn_global_load_async_to_lds_b32(
      (__attribute__((address_space(1))) int*)(unsigned long long)(const void*)g,
      (__attribute__((address_space(3))) int*)(unsigned int)(unsigned long long)(void*)l,
      0, 0);
}
#endif

__global__ void scan_kernel(const float* __restrict__ co, const float* __restrict__ dts,
                            const float* __restrict__ dAb, const float* __restrict__ Dv,
                            float* __restrict__ y) {
  const int b   = blockIdx.x >> 5;
  const int h   = blockIdx.x & 31;
  const int tid = threadIdx.x;
  const int p   = tid >> 2;
  const int ng  = tid & 3;
  const int nb  = ng * 32;

  __shared__ float sB[DSTATE_], sC[DSTATE_], sx[HEADDIM_];
  __shared__ float sdt, sdA;

  float hreg[32];
#pragma unroll
  for (int i = 0; i < 32; ++i) hreg[i] = 0.f;
  const float Dh = Dv[h];

  for (int l = 0; l < SEQLEN_; ++l) {
    const size_t row = (size_t)(b * SEQLEN_ + l);
    const float* crow = co + row * CONVDIM_;
#ifdef ASYNC_LDS_
    if (tid < 128) async_g2l_b32(crow + DINNER_ + tid, &sB[tid]);
    else           async_g2l_b32(crow + DINNER_ + DSTATE_ + (tid - 128), &sC[tid - 128]);
    if (tid < 64)  async_g2l_b32(crow + h * HEADDIM_ + tid, &sx[tid]);
    if (tid == 0) { sdt = dts[row * NHEADS_ + h]; sdA = dAb[row * NHEADS_ + h]; }
    __builtin_amdgcn_s_wait_asynccnt(0);
    __syncthreads();
#else
    if (tid < 128) sB[tid]       = crow[DINNER_ + tid];
    else           sC[tid - 128] = crow[DINNER_ + DSTATE_ + (tid - 128)];
    if (tid < 64) sx[tid] = crow[h * HEADDIM_ + tid];
    if (tid == 0) { sdt = dts[row * NHEADS_ + h]; sdA = dAb[row * NHEADS_ + h]; }
    __syncthreads();
#endif

    const float a    = sdA;
    const float coef = sdt * sx[p];
    float ysum = 0.f;
#pragma unroll
    for (int i = 0; i < 32; ++i) {
      hreg[i] = hreg[i] * a + coef * sB[nb + i];
      ysum += hreg[i] * sC[nb + i];
    }
    ysum += __shfl_xor(ysum, 1, 32);
    ysum += __shfl_xor(ysum, 2, 32);
    if (ng == 0) y[row * DINNER_ + h * HEADDIM_ + p] = ysum + Dh * sx[p];
    __syncthreads();
  }
}

// ---------------- gated RMSNorm -> f16 ----------------
__global__ void norm_kernel(const float* __restrict__ y, const float* __restrict__ zx,
                            const float* __restrict__ nw, _Float16* __restrict__ yh) {
  const int row = blockIdx.x;
  const int tid = threadIdx.x;  // 256
  __shared__ float red[256];
  float g[8];
  float ss = 0.f;
#pragma unroll
  for (int j = 0; j < 8; ++j) {
    int i = tid + j * 256;
    float gv = y[(size_t)row * DINNER_ + i] * silu_f(zx[(size_t)row * NPAD_ + i]);
    g[j] = gv; ss += gv * gv;
  }
  red[tid] = ss; __syncthreads();
  for (int s = 128; s > 0; s >>= 1) { if (tid < s) red[tid] += red[tid + s]; __syncthreads(); }
  const float scale = rsqrtf(red[0] * (1.0f / DINNER_) + 1e-5f);
#pragma unroll
  for (int j = 0; j < 8; ++j) {
    int i = tid + j * 256;
    yh[(size_t)row * DINNER_ + i] = (_Float16)(g[j] * scale * nw[i]);
  }
}

extern "C" void kernel_launch(void* const* d_in, const int* in_sizes, int n_in,
                              void* d_out, int out_size, void* d_ws, size_t ws_size,
                              hipStream_t stream) {
  (void)in_sizes; (void)n_in; (void)out_size; (void)ws_size;
  const float* x       = (const float*)d_in[0];
  const float* w_in    = (const float*)d_in[1];
  const float* conv_w  = (const float*)d_in[2];
  const float* conv_b  = (const float*)d_in[3];
  const float* dt_bias = (const float*)d_in[4];
  const float* A_log   = (const float*)d_in[5];
  const float* Dv      = (const float*)d_in[6];
  const float* norm_w  = (const float*)d_in[7];
  const float* w_out   = (const float*)d_in[8];
  float* out = (float*)d_out;

  char* base = (char*)d_ws;
  size_t off = 0;
  auto carve = [&](size_t bytes) -> void* {
    void* p = base + off;
    off += (bytes + 255) & ~(size_t)255;
    return p;
  };

  _Float16* x_h    = (_Float16*)carve((size_t)TOK_ * DMODEL_ * 2);
  _Float16* win_h  = (_Float16*)carve((size_t)NPAD_ * DMODEL_ * 2);
  float*    zx     = (float*)   carve((size_t)TOK_ * NPAD_ * 4);
  float*    co     = (float*)   carve((size_t)TOK_ * CONVDIM_ * 4);
  float*    dts    = (float*)   carve((size_t)TOK_ * NHEADS_ * 4);
  float*    dAb    = (float*)   carve((size_t)TOK_ * NHEADS_ * 4);
  float*    ybuf   = (float*)   carve((size_t)TOK_ * DINNER_ * 4);
  _Float16* yh     = (_Float16*)carve((size_t)TOK_ * DINNER_ * 2);
  _Float16* wout_h = (_Float16*)carve((size_t)DMODEL_ * DINNER_ * 2);

  // 1) cast activations & weights to f16 for the WMMA pipes
  cast_f32_to_f16<<<(TOK_ * DMODEL_ + 255) / 256, 256, 0, stream>>>(x, x_h, TOK_ * DMODEL_);
  cast_w_pad<<<(NPAD_ * DMODEL_ + 255) / 256, 256, 0, stream>>>(w_in, win_h, DINPROJ_, NPAD_, DMODEL_);
  cast_f32_to_f16<<<(DMODEL_ * DINNER_ + 255) / 256, 256, 0, stream>>>(w_out, wout_h, DMODEL_ * DINNER_);

  // 2) in_proj GEMM: zx[2048, 4416] = x_h * win_h^T
  gemm_f16_wmma<<<dim3(NPAD_ / 64, TOK_ / 32), 32, 0, stream>>>(x_h, win_h, zx, nullptr, DMODEL_, NPAD_);

  // 3) dt/dA, causal conv+SiLU
  dt_kernel<<<(TOK_ * NHEADS_ + 255) / 256, 256, 0, stream>>>(zx, dt_bias, A_log, dts, dAb);
  conv_kernel<<<(TOK_ * CONVDIM_ + 255) / 256, 256, 0, stream>>>(zx, conv_w, conv_b, co);

  // 4) selective scan (64 blocks = batch*heads)
  scan_kernel<<<BATCH_ * NHEADS_, 256, 0, stream>>>(co, dts, dAb, Dv, ybuf);

  // 5) gated RMSNorm -> f16
  norm_kernel<<<TOK_, 256, 0, stream>>>(ybuf, zx, norm_w, yh);

  // 6) out_proj GEMM with residual: out = x + yh * wout_h^T
  gemm_f16_wmma<<<dim3(DMODEL_ / 64, TOK_ / 32), 32, 0, stream>>>(yh, wout_h, out, x, DINNER_, DMODEL_);
}
